// Lamm_27685359190625
// MI455X (gfx1250) — matrix operand order repaired and checked by reading, not compiled, not run on hardware
//
#include <hip/hip_runtime.h>

typedef __attribute__((ext_vector_type(16))) _Float16 v16h;
typedef __attribute__((ext_vector_type(8)))  float    v8f;

// ---------------------------------------------------------------------------
// Zero the small accumulator workspace (10 floats: sum_h[5], cover[5]).
// ---------------------------------------------------------------------------
__global__ void zero_ws_kernel(float* __restrict__ ws) {
    if (threadIdx.x < 16) ws[threadIdx.x] = 0.0f;
}

// ---------------------------------------------------------------------------
// Bandwidth-bound reduction: sum of one level's activations into ws slot.
// ---------------------------------------------------------------------------
__global__ void sum_kernel(const float* __restrict__ x, long long n,
                           float* __restrict__ acc) {
    float s = 0.0f;
    long long i      = (long long)blockIdx.x * blockDim.x + threadIdx.x;
    long long stride = (long long)gridDim.x * blockDim.x;
    for (; i < n; i += stride) s += x[i];

    // wave32 reduce
    #pragma unroll
    for (int off = 16; off > 0; off >>= 1) s += __shfl_down(s, off, 32);

    __shared__ float ls[8];
    const int wid = threadIdx.x >> 5;
    const int lid = threadIdx.x & 31;
    if (lid == 0) ls[wid] = s;
    __syncthreads();
    if (threadIdx.x == 0) {
        float t = 0.0f;
        const int nw = blockDim.x >> 5;
        for (int w = 0; w < nw; ++w) t += ls[w];
        atomicAdd(acc, t);
    }
}

// ---------------------------------------------------------------------------
// WMMA coverage kernel: per (batch, 16x16 tile) compute
//   cover[h,w] = sum_m row[m,h]*col[m,w]   (K = 64 boxes, 2x 16x16x32 WMMA)
// then count cover > 0 and accumulate per-level covered-pixel count.
// grid.y = batch, each wave of the 256-thread block owns one 16x16 tile.
// ---------------------------------------------------------------------------
__global__ void cover_kernel(const float* __restrict__ boxes,
                             const int*  __restrict__ dimx_p,
                             const int*  __restrict__ dimy_p,
                             int H, int W, int tiles_x, int num_tiles,
                             float* __restrict__ acc) {
    __shared__ float2 sy[64];   // (y1, y2) per box, valid folded into y
    __shared__ float2 sx[64];   // (x1, x2) per box
    __shared__ int    blk_count;

    const int b   = blockIdx.y;
    const int tid = threadIdx.x;
    if (tid == 0) blk_count = 0;

    if (tid < 64) {
        const float* bp = boxes + ((long long)b * 64 + tid) * 4;
        const float bx1 = bp[0], by1 = bp[1], bx2 = bp[2], by2 = bp[3];
        const float scx = (float)W / (float)dimx_p[0];
        const float scy = (float)H / (float)dimy_p[0];
        // jnp.round == rint (round half to even), then clip — as in reference
        float x1 = fminf(fmaxf(rintf(bx1 * scx), 0.0f), (float)(W - 1));
        float y1 = fminf(fmaxf(rintf(by1 * scy), 0.0f), (float)(H - 1));
        float x2 = fminf(fmaxf(rintf(bx2 * scx), 0.0f), (float)W);
        float y2 = fminf(fmaxf(rintf(by2 * scy), 0.0f), (float)H);
        const bool valid = (x2 > x1) && (y2 > y1);
        if (!valid) { y1 = 0.0f; y2 = 0.0f; }  // empty row interval kills box
        sy[tid] = make_float2(y1, y2);
        sx[tid] = make_float2(x1, x2);
    }
    __syncthreads();

    const int wave = tid >> 5;     // 8 waves per block, one tile each
    const int lane = tid & 31;
    const int t    = blockIdx.x * 8 + wave;   // wave-uniform: EXEC stays full

    if (t < num_tiles) {
        const int th0 = (t / tiles_x) << 4;
        const int tw0 = (t % tiles_x) << 4;
        const int g   = lane >> 4;            // lane half-group
        const int l15 = lane & 15;
        const float hf = (float)(th0 + l15);  // A-matrix row M = lane&15
        const float wf = (float)(tw0 + l15);  // B-matrix col N = lane&15

        const _Float16 one  = (_Float16)1.0f;
        const _Float16 zero = (_Float16)0.0f;

        v8f c = {};
        #pragma unroll
        for (int kb = 0; kb < 64; kb += 32) {
            // Phase 1: unconditional clustered LDS loads (ds_load_b64 each)
            float2 ya[16], xb[16];
            #pragma unroll
            for (int e = 0; e < 16; ++e) {
                // A 16x32 f16 layout: K = 16*g + e
                const int ma = kb + (g << 4) + e;
                // B 32x16 f16 layout: K = 16*g + e
                const int mb = kb + (g << 4) + e;
                ya[e] = sy[ma];
                xb[e] = sx[mb];
            }
            // Phase 2: branchless indicator build (v_cndmask_b16 per element)
            v16h a, bm;
            #pragma unroll
            for (int e = 0; e < 16; ++e) {
                const int ca = (int)(hf >= ya[e].x) & (int)(hf < ya[e].y);
                const int cb = (int)(wf >= xb[e].x) & (int)(wf < xb[e].y);
                a[e]  = ca ? one : zero;
                bm[e] = cb ? one : zero;
            }
            // D = A x B + C  (v_wmma_f32_16x16x32_f16)
            c = __builtin_amdgcn_wmma_f32_16x16x32_f16(
                    false, a, false, bm, (short)0, c, false, false);
        }

        // C/D layout: element r at lane -> (M = r + 8*g, N = lane&15)
        int cnt = 0;
        #pragma unroll
        for (int r = 0; r < 8; ++r) {
            const int h   = th0 + r + (g << 3);
            const int w   = tw0 + l15;
            const int inb = (int)(h < H) & (int)(w < W) & (int)(c[r] > 0.5f);
            cnt += inb;
        }
        #pragma unroll
        for (int off = 16; off > 0; off >>= 1) cnt += __shfl_down(cnt, off, 32);
        if (lane == 0) atomicAdd(&blk_count, cnt);
    }
    __syncthreads();
    if (tid == 0 && blk_count != 0) atomicAdd(acc, (float)blk_count);
}

// ---------------------------------------------------------------------------
// Combine: loss = mean_l ((sum_h[l] - cover[l]) / tn[l])^2
// ---------------------------------------------------------------------------
__global__ void final_kernel(const float* __restrict__ ws, float* __restrict__ out,
                             float tn0, float tn1, float tn2, float tn3, float tn4) {
    if (threadIdx.x == 0 && blockIdx.x == 0) {
        const float tn[5] = {tn0, tn1, tn2, tn3, tn4};
        float loss = 0.0f;
        #pragma unroll
        for (int l = 0; l < 5; ++l) {
            const float d = (ws[l] - ws[5 + l]) / tn[l];
            loss += d * d;
        }
        out[0] = loss * 0.2f;
    }
}

// ---------------------------------------------------------------------------
extern "C" void kernel_launch(void* const* d_in, const int* in_sizes, int n_in,
                              void* d_out, int out_size, void* d_ws, size_t ws_size,
                              hipStream_t stream) {
    (void)in_sizes; (void)n_in; (void)out_size; (void)ws_size;
    static const int Hs[5] = {334, 167, 84, 42, 21};
    static const int Ws[5] = {200, 100, 50, 25, 13};

    const float* boxes = (const float*)d_in[5];
    const int*   dimx  = (const int*)d_in[6];
    const int*   dimy  = (const int*)d_in[7];
    float*       ws    = (float*)d_ws;   // [0..4]=sum_h, [5..9]=cover count

    zero_ws_kernel<<<1, 32, 0, stream>>>(ws);

    for (int l = 0; l < 5; ++l) {
        const long long n = 128LL * Hs[l] * Ws[l];
        int blocks = (int)((n + 255) / 256);
        if (blocks > 1024) blocks = 1024;
        sum_kernel<<<blocks, 256, 0, stream>>>((const float*)d_in[l], n, ws + l);

        const int tx = (Ws[l] + 15) / 16;
        const int ty = (Hs[l] + 15) / 16;
        const int nt = tx * ty;
        dim3 grid((nt + 7) / 8, 128);
        cover_kernel<<<grid, 256, 0, stream>>>(boxes, dimx, dimy,
                                               Hs[l], Ws[l], tx, nt, ws + 5 + l);
    }

    final_kernel<<<1, 32, 0, stream>>>(ws, (float*)d_out,
        128.0f * 334 * 200, 128.0f * 167 * 100, 128.0f * 84 * 50,
        128.0f * 42 * 25, 128.0f * 21 * 13);
}
